// MultiHeadAttention_944892805207
// MI455X (gfx1250) — matrix operand order repaired and checked by reading, not compiled, not run on hardware
//
#include <hip/hip_runtime.h>

// MI455X / gfx1250: wave32, WMMA 16x16x32 f16 -> f32 accum.
// Data movement: TDM (tensor_load_to_lds), async global->LDS, ds staging.

typedef __attribute__((ext_vector_type(16))) _Float16 v16h;
typedef __attribute__((ext_vector_type(8)))  _Float16 v8h;
typedef __attribute__((ext_vector_type(8)))  float    v8f;
typedef __attribute__((ext_vector_type(4)))  float    v4f;
typedef __attribute__((ext_vector_type(4)))  unsigned v4u;
typedef __attribute__((ext_vector_type(8)))  unsigned v8u;

#define WMMA_F16(a, b, c) \
  __builtin_amdgcn_wmma_f32_16x16x32_f16(false, (a), false, (b), (short)0, (c), false, false)

union V16U { v16h v; v8h h[2]; };
static __device__ __forceinline__ v16h make_frag(v8h lo, v8h hi) {
  V16U u; u.h[0] = lo; u.h[1] = hi; return u.v;
}

// ---- DPP16 butterfly reductions within each 16-lane half-row --------------
template <int CTRL>
static __device__ __forceinline__ float dppf(float x) {
  return __int_as_float(__builtin_amdgcn_update_dpp(
      0, __float_as_int(x), CTRL, 0xf, 0xf, true));
}
static __device__ __forceinline__ float red_max16(float x) {
  x = fmaxf(x, dppf<0xB1>(x));    // quad_perm [1,0,3,2]  (xor 1)
  x = fmaxf(x, dppf<0x4E>(x));    // quad_perm [2,3,0,1]  (xor 2)
  x = fmaxf(x, dppf<0x141>(x));   // row_half_mirror      (xor 4)
  x = fmaxf(x, dppf<0x140>(x));   // row_mirror           (xor 8)
  return x;
}
static __device__ __forceinline__ float red_add16(float x) {
  x += dppf<0xB1>(x);
  x += dppf<0x4E>(x);
  x += dppf<0x141>(x);
  x += dppf<0x140>(x);
  return x;
}

// ---- async global -> LDS copy (16B per lane), ASYNCcnt tracked ------------
static __device__ __forceinline__ void async_copy_b128(unsigned lds_off,
                                                       const void* gaddr) {
  asm volatile("global_load_async_to_lds_b128 %0, %1, off"
               :: "v"(lds_off), "v"((unsigned long long)(uintptr_t)gaddr)
               : "memory");
}
static __device__ __forceinline__ void wait_async0() {
  asm volatile("s_wait_asynccnt 0" ::: "memory");
}
static __device__ __forceinline__ unsigned lds_off32(const void* p) {
  return (unsigned)(uintptr_t)p;   // LDS aperture: addr[31:0] is the LDS offset
}

// ---- Tensor Data Mover: 2D tile [rows x cols(f16)] global -> LDS ----------
// LDS rows padded 128B data + 16B pad (stride 144B = 72 halfs):
// pad_interval code 4 (32 DWORDs), pad_amount code 3 (4 DWORDs).
static __device__ __forceinline__ void tdm_load_tile_f16(
    unsigned lds_off, const void* gaddr, unsigned tensor_d0, unsigned tensor_d1,
    unsigned tile_d0, unsigned tile_d1, unsigned stride_d0) {
  const unsigned long long ga = (unsigned long long)(uintptr_t)gaddr;
  v4u g0;
  g0[0] = 1u;                                                 // count=1 (user D#)
  g0[1] = lds_off;                                            // lds_addr
  g0[2] = (unsigned)ga;                                       // global_addr[31:0]
  g0[3] = (unsigned)((ga >> 32) & 0x01FFFFFFu) | (2u << 30);  // ga[56:32], type=2
  v8u g1;
  g1[0] = (1u << 16) | (1u << 20) | (4u << 22) | (3u << 25);  // 2B elems, padding
  g1[1] = (tensor_d0 & 0xFFFFu) << 16;                        // tensor_dim0 lo16
  g1[2] = ((tensor_d0 >> 16) & 0xFFFFu) | ((tensor_d1 & 0xFFFFu) << 16);
  g1[3] = ((tensor_d1 >> 16) & 0xFFFFu) | ((tile_d0 & 0xFFFFu) << 16);
  g1[4] = tile_d1 & 0xFFFFu;                                  // tile_dim1 (dim2=0)
  g1[5] = stride_d0;                                          // dim0_stride[31:0]
  g1[6] = 0u;                                                 // stride hi / dim1_stride
  g1[7] = 0u;
  v4u gz = {0u, 0u, 0u, 0u};
  asm volatile("tensor_load_to_lds %0, %1, %2, %3"
               :: "s"(g0), "s"(g1), "s"(gz), "s"(gz)
               : "memory");
}

// ---------------------------------------------------------------------------
// GEMM: Y[M,N] = X[M,K] @ W[N,K]^T + bias[N]
// Block 256 thr = 8 waves. Tile 128(M) x 64(N), K-step 64 -> 8 WMMA/wave/iter.
// ---------------------------------------------------------------------------
template <bool XF16, bool OUTF32>
__global__ __launch_bounds__(256) void gemm_xwT(
    const void* __restrict__ Xv, const float* __restrict__ W,
    const float* __restrict__ bias, void* __restrict__ Yv,
    int M, int N, int K)
{
  constexpr int BM = 128, BN = 64, BK = 64;
  constexpr int LDA = 72, LDB = 72;            // halfs; 144 B rows (16B multiple)
  __shared__ _Float16 Xs[BM * LDA];
  __shared__ _Float16 Ws[BN * LDB];

  const int t    = threadIdx.x;
  const int lane = t & 31;
  const int wave = t >> 5;
  const int half = lane >> 4;
  const int l15  = lane & 15;

  const int mBlk = blockIdx.x * BM;
  const int nBlk = blockIdx.y * BN;

  const float*    Xf = (const float*)Xv;
  const _Float16* Xh = (const _Float16*)Xv;

  v8f acc[4] = {};

  for (int k0 = 0; k0 < K; k0 += BK) {
    // ---- stage X tile (128 x 64) ----
    if constexpr (!XF16) {
      #pragma unroll
      for (int i = 0; i < 8; ++i) {             // 2048 float4 chunks / 256 thr
        int c = t + i * 256;
        int row = c >> 4, c4 = c & 15;
        v4f x = *(const v4f*)&Xf[(size_t)(mBlk + row) * K + k0 + c4 * 4];
        _Float16* d = &Xs[row * LDA + c4 * 4];
        d[0] = (_Float16)x[0]; d[1] = (_Float16)x[1];
        d[2] = (_Float16)x[2]; d[3] = (_Float16)x[3];
      }
    } else {
      #pragma unroll
      for (int i = 0; i < 4; ++i) {             // 1024 b128 chunks, async DMA
        int c = t + i * 256;
        int row = c >> 3, c8 = c & 7;
        async_copy_b128(lds_off32(&Xs[row * LDA + c8 * 8]),
                        &Xh[(size_t)(mBlk + row) * K + k0 + c8 * 8]);
      }
    }
    // ---- stage W tile (64 x 64), f32 -> f16 ----
    #pragma unroll
    for (int i = 0; i < 4; ++i) {               // 1024 float4 chunks / 256 thr
      int c = t + i * 256;
      int row = c >> 4, c4 = c & 15;
      v4f w = *(const v4f*)&W[(size_t)(nBlk + row) * K + k0 + c4 * 4];
      _Float16* d = &Ws[row * LDB + c4 * 4];
      d[0] = (_Float16)w[0]; d[1] = (_Float16)w[1];
      d[2] = (_Float16)w[2]; d[3] = (_Float16)w[3];
    }
    if (k0 + BK < K) {
      __builtin_prefetch(&W[(size_t)nBlk * K + k0 + BK], 0, 1);
      if constexpr (!XF16) __builtin_prefetch(&Xf[(size_t)mBlk * K + k0 + BK], 0, 1);
    }
    if constexpr (XF16) wait_async0();
    __syncthreads();

    // A fragments (16x32 each): lanes 0-15 M=l15 K{0..7,16..23}; 16-31 K{8..15,24..31}
    const int mRow = wave * 16 + l15;
    v16h A0 = make_frag(*(const v8h*)&Xs[mRow * LDA +      half * 8],
                        *(const v8h*)&Xs[mRow * LDA + 16 + half * 8]);
    v16h A1 = make_frag(*(const v8h*)&Xs[mRow * LDA + 32 + half * 8],
                        *(const v8h*)&Xs[mRow * LDA + 48 + half * 8]);

    #pragma unroll
    for (int nt = 0; nt < 4; ++nt) {
      const int nRow = nt * 16 + l15;
      v16h Bl = make_frag(*(const v8h*)&Ws[nRow * LDB +      half * 8],
                          *(const v8h*)&Ws[nRow * LDB + 16 + half * 8]);
      v16h Bh = make_frag(*(const v8h*)&Ws[nRow * LDB + 32 + half * 8],
                          *(const v8h*)&Ws[nRow * LDB + 48 + half * 8]);
      acc[nt] = WMMA_F16(A0, Bl, acc[nt]);
      acc[nt] = WMMA_F16(A1, Bh, acc[nt]);
    }
    __syncthreads();
  }

  // ---- epilogue: C layout -> lanes 0-15 rows r, lanes 16-31 rows r+8 ----
  const int mBase = mBlk + wave * 16 + half * 8;
  #pragma unroll
  for (int nt = 0; nt < 4; ++nt) {
    const int n = nBlk + nt * 16 + l15;
    const float bv = bias[n];
    #pragma unroll
    for (int r = 0; r < 8; ++r) {
      float y = acc[nt][r] + bv;
      size_t off = (size_t)(mBase + r) * N + n;
      if constexpr (OUTF32) ((float*)Yv)[off] = y;
      else                  ((_Float16*)Yv)[off] = (_Float16)y;
    }
  }
}

// ---------------------------------------------------------------------------
// Flash attention: blockIdx.y = (b,h); 128 queries/block, 8 waves x 16 queries.
// DK = 64, key tiles of 64, online softmax in base-2 domain.
// K tile via one TDM descriptor (padded rows); V staged transposed;
// P transposed to A-layout through per-wave LDS buffer.
// ---------------------------------------------------------------------------
__global__ __launch_bounds__(256) void flash_attn(
    const _Float16* __restrict__ Qp, const _Float16* __restrict__ Kp,
    const _Float16* __restrict__ Vp, _Float16* __restrict__ Out,
    int S, int D, int H)
{
  constexpr int DK  = 64;
  constexpr int LKS = 72, LVT = 72, LPS = 72;   // half strides, 144 B rows
  __shared__ _Float16 Ks[64 * LKS];             // K tile   [key][dk]
  __shared__ _Float16 Vt[DK * LVT];             // V tile^T [dk][key]
  __shared__ _Float16 Ps[8 * 16 * LPS];         // per-wave P staging [row][key]

  const int t    = threadIdx.x;
  const int lane = t & 31;
  const int wave = t >> 5;
  const int half = lane >> 4;
  const int l15  = lane & 15;

  const int bh = blockIdx.y;
  const int b  = bh / H, h = bh % H;
  const int d0 = h * DK;
  const int q0 = blockIdx.x * 128 + wave * 16;

  const float SCALE = 0.125f * 1.44269504088896f;  // (1/sqrt(DK)) * log2(e)

  // Q A-fragments straight from global (16B-aligned contiguous runs).
  const size_t qrow = (size_t)(b * S + q0 + l15) * D + d0;
  const v16h A0 = make_frag(*(const v8h*)&Qp[qrow +      half * 8],
                            *(const v8h*)&Qp[qrow + 16 + half * 8]);
  const v16h A1 = make_frag(*(const v8h*)&Qp[qrow + 32 + half * 8],
                            *(const v8h*)&Qp[qrow + 48 + half * 8]);

  v8f O0 = {}, O1 = {}, O2 = {}, O3 = {};
  float Mx[8], L[8];
  #pragma unroll
  for (int r = 0; r < 8; ++r) { Mx[r] = -1e30f; L[r] = 0.f; }

  _Float16* myPs = &Ps[wave * 16 * LPS];

  for (int k0 = 0; k0 < S; k0 += 64) {
    // ---- K tile [64 keys x 64 dk] : one Tensor Data Mover descriptor ----
    if (wave == 0) {
      tdm_load_tile_f16(lds_off32(Ks),
                        &Kp[(size_t)(b * S + k0) * D + d0],
                        /*tensor_d0=*/DK, /*tensor_d1=*/(unsigned)S,
                        /*tile_d0=*/DK,   /*tile_d1=*/64u,
                        /*stride_d0=*/(unsigned)D);
    }
    // ---- V tile transposed [dk][key] (load v8h, scatter 8 halfs) ----
    #pragma unroll
    for (int i = 0; i < 2; ++i) {
      int c = t + i * 256;
      int row = c >> 3, c8 = c & 7;
      v8h vv = *(const v8h*)&Vp[(size_t)(b * S + k0 + row) * D + d0 + c8 * 8];
      #pragma unroll
      for (int j = 0; j < 8; ++j) Vt[(c8 * 8 + j) * LVT + row] = vv[j];
    }
    if (wave == 0) __builtin_amdgcn_s_wait_tensorcnt(0);
    __syncthreads();

    // ---- scores 16x64 : four 16x16 f32 accumulators, 8 WMMA ----
    v8f c0 = {}, c1 = {}, c2 = {}, c3 = {};
    #pragma unroll
    for (int nt = 0; nt < 4; ++nt) {
      const int key = nt * 16 + l15;
      v16h Bl = make_frag(*(const v8h*)&Ks[key * LKS +      half * 8],
                          *(const v8h*)&Ks[key * LKS + 16 + half * 8]);
      v16h Bh = make_frag(*(const v8h*)&Ks[key * LKS + 32 + half * 8],
                          *(const v8h*)&Ks[key * LKS + 48 + half * 8]);
      v8f c = {};
      c = WMMA_F16(A0, Bl, c);
      c = WMMA_F16(A1, Bh, c);
      if      (nt == 0) c0 = c;
      else if (nt == 1) c1 = c;
      else if (nt == 2) c2 = c;
      else              c3 = c;
    }

    // ---- online softmax; rows r + 8*half; DPP16 reductions, no LDS ----
    float scale_old[8];
    #pragma unroll
    for (int r = 0; r < 8; ++r) {
      float s0 = c0[r] * SCALE, s1 = c1[r] * SCALE;
      float s2 = c2[r] * SCALE, s3 = c3[r] * SCALE;
      c0[r] = s0; c1[r] = s1; c2[r] = s2; c3[r] = s3;
      float m = red_max16(fmaxf(fmaxf(s0, s1), fmaxf(s2, s3)));
      float mnew = fmaxf(Mx[r], m);
      scale_old[r] = __builtin_exp2f(Mx[r] - mnew);
      Mx[r] = mnew;
    }
    #pragma unroll
    for (int r = 0; r < 8; ++r) {
      float p0 = __builtin_exp2f(c0[r] - Mx[r]);
      float p1 = __builtin_exp2f(c1[r] - Mx[r]);
      float p2 = __builtin_exp2f(c2[r] - Mx[r]);
      float p3 = __builtin_exp2f(c3[r] - Mx[r]);
      float ps = red_add16(p0 + p1 + p2 + p3);
      L[r] = L[r] * scale_old[r] + ps;
      O0[r] *= scale_old[r]; O1[r] *= scale_old[r];
      O2[r] *= scale_old[r]; O3[r] *= scale_old[r];
      // C-layout -> A-layout transpose via per-wave LDS buffer
      const int row = r + 8 * half;
      myPs[row * LPS +      l15] = (_Float16)p0;
      myPs[row * LPS + 16 + l15] = (_Float16)p1;
      myPs[row * LPS + 32 + l15] = (_Float16)p2;
      myPs[row * LPS + 48 + l15] = (_Float16)p3;
    }
    asm volatile("s_wait_dscnt 0" ::: "memory");   // intra-wave LDS RAW

    // ---- O += P V : two P A-fragments (keys 0-31, 32-63), 8 WMMA ----
    {
      v16h P0 = make_frag(*(const v8h*)&myPs[l15 * LPS +      half * 8],
                          *(const v8h*)&myPs[l15 * LPS + 16 + half * 8]);
      v16h P1 = make_frag(*(const v8h*)&myPs[l15 * LPS + 32 + half * 8],
                          *(const v8h*)&myPs[l15 * LPS + 48 + half * 8]);
      #pragma unroll
      for (int nt = 0; nt < 4; ++nt) {
        const int dk = nt * 16 + l15;
        v16h Bv0 = make_frag(*(const v8h*)&Vt[dk * LVT +      half * 8],
                             *(const v8h*)&Vt[dk * LVT + 16 + half * 8]);
        v16h Bv1 = make_frag(*(const v8h*)&Vt[dk * LVT + 32 + half * 8],
                             *(const v8h*)&Vt[dk * LVT + 48 + half * 8]);
        v8f* Op = (nt == 0) ? &O0 : (nt == 1) ? &O1 : (nt == 2) ? &O2 : &O3;
        *Op = WMMA_F16(P0, Bv0, *Op);
        *Op = WMMA_F16(P1, Bv1, *Op);
      }
    }
    __syncthreads();   // protect Ks/Vt before next tile's staging
  }

  // ---- normalize and emit concat layout [B,S,D] in f16 ----
  #pragma unroll
  for (int r = 0; r < 8; ++r) {
    const float inv = 1.0f / L[r];
    const int m = q0 + r + 8 * half;
    const size_t base = (size_t)(b * S + m) * D + d0;
    Out[base +  0 + l15] = (_Float16)(O0[r] * inv);
    Out[base + 16 + l15] = (_Float16)(O1[r] * inv);
    Out[base + 32 + l15] = (_Float16)(O2[r] * inv);
    Out[base + 48 + l15] = (_Float16)(O3[r] * inv);
  }
}

// ---------------------------------------------------------------------------
extern "C" void kernel_launch(void* const* d_in, const int* in_sizes, int n_in,
                              void* d_out, int out_size, void* d_ws, size_t ws_size,
                              hipStream_t stream) {
  const float* q  = (const float*)d_in[0];
  const float* k  = (const float*)d_in[1];
  const float* v  = (const float*)d_in[2];
  const float* Wq = (const float*)d_in[3];
  const float* bq = (const float*)d_in[4];
  const float* Wk = (const float*)d_in[5];
  const float* bk = (const float*)d_in[6];
  const float* Wv = (const float*)d_in[7];
  const float* bv = (const float*)d_in[8];
  const float* Wo = (const float*)d_in[9];
  const float* bo = (const float*)d_in[10];

  const int B = 2, S = 2048, D = 1024, H = 16;
  const int M = B * S;

  _Float16* Qp = (_Float16*)d_ws;                 // [M, D] f16
  _Float16* Kp = Qp + (size_t)M * D;              // [M, D] f16
  _Float16* Vp = Kp + (size_t)M * D;              // [M, D] f16
  _Float16* Ao = Vp + (size_t)M * D;              // [M, D] f16 (attn concat)

  dim3 blk(256);
  dim3 gGemm(M / 128, D / 64);
  hipLaunchKernelGGL((gemm_xwT<false, false>), gGemm, blk, 0, stream,
                     (const void*)q, Wq, bq, (void*)Qp, M, D, D);
  hipLaunchKernelGGL((gemm_xwT<false, false>), gGemm, blk, 0, stream,
                     (const void*)k, Wk, bk, (void*)Kp, M, D, D);
  hipLaunchKernelGGL((gemm_xwT<false, false>), gGemm, blk, 0, stream,
                     (const void*)v, Wv, bv, (void*)Vp, M, D, D);

  dim3 gAttn(S / 128, B * H);
  hipLaunchKernelGGL(flash_attn, gAttn, blk, 0, stream, Qp, Kp, Vp, Ao, S, D, H);

  hipLaunchKernelGGL((gemm_xwT<true, true>), gGemm, blk, 0, stream,
                     (const void*)Ao, Wo, bo, d_out, M, D, D);
}
